// CompressingBlock_17970143166606
// MI455X (gfx1250) — matrix operand order repaired and checked by reading, not compiled, not run on hardware
//
#include <hip/hip_runtime.h>
#include <stdint.h>

#define DEV static __device__ __forceinline__

typedef __attribute__((ext_vector_type(16))) __bf16 v16bf;
typedef __attribute__((ext_vector_type(8)))  __bf16 v8bf;
typedef __attribute__((ext_vector_type(8)))  float  v8f;
typedef __attribute__((address_space(3)))    uint16_t lds16_t;

// Problem dims (fixed by the reference)
static constexpr int Bx  = 8, Tx = 1024, TRx = 256, Dx = 1024, Hx = 16, HSx = 64, FFx = 4096;
static constexpr int BT  = Bx * Tx;    // 8192 rows
static constexpr int BTR = Bx * TRx;   // 2048 rows

// ---------------- helpers ----------------
DEV uint16_t f2bf(float f) {
  uint32_t u = __builtin_bit_cast(uint32_t, f);
  uint32_t r = u + 0x7FFFu + ((u >> 16) & 1u);       // round-to-nearest-even
  if ((u & 0x7F800000u) == 0x7F800000u) r = u;       // keep inf/nan
  return (uint16_t)(r >> 16);
}

// Build a v16bf A/B fragment from two contiguous 8-element (16B) chunks.
DEV v16bf frag(const uint16_t* p0, const uint16_t* p1) {
  v8bf lo = *(const v8bf*)p0;
  v8bf hi = *(const v8bf*)p1;
  return __builtin_shufflevector(lo, hi, 0,1,2,3,4,5,6,7,8,9,10,11,12,13,14,15);
}

DEV v8f wmma_bf16(v16bf a, v16bf b, v8f c) {
  return __builtin_amdgcn_wmma_f32_16x16x32_bf16(false, a, false, b, (short)0, c, false, false);
}

// gfx1250 async global->LDS copy (ISA §15.18.3 op 98, tracked by ASYNCcnt).
// Per-lane: copies 16 bytes from global vaddr to LDS address vdst, no VGPR data round-trip.
DEV uint32_t lds_off32(uint16_t* p) {
  return (uint32_t)(uintptr_t)(lds16_t*)p;
}
DEV void async_ld_b128(uint16_t* lds_dst, const uint16_t* gsrc) {
  asm volatile("global_load_async_to_lds_b128 %0, %1, off"
               :: "v"(lds_off32(lds_dst)), "v"(gsrc) : "memory");
}
DEV void wait_async0() { asm volatile("s_wait_asynccnt 0" ::: "memory"); }
DEV void wait_ds0()    { asm volatile("s_wait_dscnt 0" ::: "memory"); }

// ---------------- weight conversion ----------------
// (H, D, HS) fp32 -> bf16 Wt[n=h*HS+s][k=c]   (pre-transposed for B-fragment loads)
__global__ __launch_bounds__(256) void cvt_qkv_w(const float* __restrict__ w,
                                                 uint16_t* __restrict__ out) {
  int idx = blockIdx.x * 256 + threadIdx.x;   // over D*D, idx = n*D + c
  int n = idx / Dx, c = idx % Dx;
  int h = n / HSx, s = n % HSx;
  out[idx] = f2bf(w[((size_t)h * Dx + c) * HSx + s]);
}

// [K,N] fp32 -> bf16 [N,K]
__global__ __launch_bounds__(256) void transpose_w(const float* __restrict__ in,
                                                   uint16_t* __restrict__ out, int K, int N) {
  __shared__ float t[32][33];
  const int k0 = blockIdx.x * 32, n0 = blockIdx.y * 32;
  const int tx = threadIdx.x & 31, ty = threadIdx.x >> 5;
  for (int r = ty; r < 32; r += 8) t[r][tx] = in[(size_t)(k0 + r) * N + n0 + tx];
  __syncthreads();
  for (int r = ty; r < 32; r += 8) out[(size_t)(n0 + r) * K + k0 + tx] = f2bf(t[tx][r]);
}

__global__ __launch_bounds__(256) void cvt_f2bf(const float* __restrict__ in,
                                                uint16_t* __restrict__ out, int n) {
  int i = blockIdx.x * 256 + threadIdx.x;
  if (i < n) out[i] = f2bf(in[i]);
}

__global__ __launch_bounds__(256) void add_inplace(float* __restrict__ y,
                                                   const float* __restrict__ x, int n) {
  int i = blockIdx.x * 256 + threadIdx.x;
  if (i < n) y[i] += x[i];
}

// ---------------- fused LayerNorm -> bf16 (D = 1024, 256 threads, 4 elems/thread) --------------
__global__ __launch_bounds__(256) void layernorm_bf16_k(const float* __restrict__ X,
                                                        const float* __restrict__ g,
                                                        const float* __restrict__ be,
                                                        uint16_t* __restrict__ out) {
  constexpr int D = Dx;
  const int row = blockIdx.x, tid = threadIdx.x;
  const float* xr = X + (size_t)row * D;
  float v[4]; float s = 0.f, sq = 0.f;
  #pragma unroll
  for (int i = 0; i < 4; i++) { v[i] = xr[tid + 256 * i]; s += v[i]; sq += v[i] * v[i]; }
  #pragma unroll
  for (int o = 16; o > 0; o >>= 1) { s += __shfl_xor(s, o, 32); sq += __shfl_xor(sq, o, 32); }
  __shared__ float ws[8], wq[8];
  if ((tid & 31) == 0) { ws[tid >> 5] = s; wq[tid >> 5] = sq; }
  __syncthreads();
  s = 0.f; sq = 0.f;
  #pragma unroll
  for (int w = 0; w < 8; w++) { s += ws[w]; sq += wq[w]; }
  const float mean = s * (1.f / D);
  const float var  = sq * (1.f / D) - mean * mean;
  const float rstd = rsqrtf(var + 1e-5f);
  uint16_t* orow = out + (size_t)row * D;
  #pragma unroll
  for (int i = 0; i < 4; i++) { int c = tid + 256 * i;
    orow[c] = f2bf((v[i] - mean) * rstd * g[c] + be[c]); }
}

// ---------------- WMMA GEMM: out[M,N] = A[M,K](bf16) * Wt[N,K](bf16) + epilogue --------------
// Block = 256 thr = 8 waves; tile 128x128x32; wave computes 32x64 (2x4 WMMA tiles).
// Double-buffered LDS; next k-tile copied with GLOBAL_LOAD_ASYNC_TO_LDS_B128 while the
// current tile feeds the WMMA pipe. One barrier per k-step.
template<bool RELU, bool BIAS, bool RES, bool OUTF, bool OUTB>
__global__ __launch_bounds__(256)
void gemm_bf16_k(const uint16_t* __restrict__ A, const uint16_t* __restrict__ Wt,
                 const float* __restrict__ bias, const float* __restrict__ res,
                 float* __restrict__ outF, uint16_t* __restrict__ outB,
                 int M, int N, int K) {
  constexpr int BM = 128, BN = 128, BK = 32, LDK = 40;  // LDK pad: 80B rows, 16B-aligned chunks
  __shared__ __align__(16) uint16_t As[2][BM * LDK];
  __shared__ __align__(16) uint16_t Bs[2][BN * LDK];
  const int tid = threadIdx.x;
  const int nbn = N / BN;
  const int bm = (blockIdx.x / nbn) * BM;
  const int bn = (blockIdx.x % nbn) * BN;
  const int wid = tid >> 5, lane = tid & 31, l15 = lane & 15;
  const int wm = (wid & 3) * 32, wn = (wid >> 2) * 64;
  const int half8 = (lane < 16) ? 0 : 8;
  const int abase = half8 * 2;    // A frag: contiguous K run of 16 at 0 or 16
  const int bbase = half8 * 2;    // B frag: contiguous K run of 16 at 0 or 16

  auto load_tiles = [&](int buf, int kt) {
    for (int c = tid; c < (BM * BK / 8); c += 256) {
      int row = c >> 2, kc = (c & 3) << 3;
      async_ld_b128(&As[buf][row * LDK + kc], &A[(size_t)(bm + row) * K + kt + kc]);
    }
    for (int c = tid; c < (BN * BK / 8); c += 256) {
      int row = c >> 2, kc = (c & 3) << 3;
      async_ld_b128(&Bs[buf][row * LDK + kc], &Wt[(size_t)(bn + row) * K + kt + kc]);
    }
  };

  v8f acc[2][4] = {};
  load_tiles(0, 0);
  wait_async0();
  __syncthreads();
  int cur = 0;
  for (int kt = 0; kt < K; kt += BK) {
    if (kt + BK < K) load_tiles(cur ^ 1, kt + BK);   // overlap copy with WMMA below
    v16bf af[2], bfr[4];
    #pragma unroll
    for (int i = 0; i < 2; i++) {
      const uint16_t* pr = &As[cur][(wm + i * 16 + l15) * LDK];
      af[i] = frag(pr + abase, pr + abase + 8);
    }
    #pragma unroll
    for (int j = 0; j < 4; j++) {
      const uint16_t* pr = &Bs[cur][(wn + j * 16 + l15) * LDK];
      bfr[j] = frag(pr + bbase, pr + bbase + 8);
    }
    #pragma unroll
    for (int i = 0; i < 2; i++)
      #pragma unroll
      for (int j = 0; j < 4; j++)
        acc[i][j] = wmma_bf16(af[i], bfr[j], acc[i][j]);
    wait_async0();
    __syncthreads();
    cur ^= 1;
  }
  // Epilogue. C layout: N = lane&15, M = vgpr_idx + (lane<16 ? 0 : 8)
  #pragma unroll
  for (int i = 0; i < 2; i++) {
    #pragma unroll
    for (int j = 0; j < 4; j++) {
      const int n = bn + wn + j * 16 + l15;
      #pragma unroll
      for (int r = 0; r < 8; r++) {
        const int m = bm + wm + i * 16 + r + half8;
        float v = acc[i][j][r];
        if (BIAS) v += bias[n];
        if (RES)  v += res[(size_t)m * N + n];
        if (RELU) v = fmaxf(v, 0.f);
        if (OUTF) outF[(size_t)m * N + n] = v;
        if (OUTB) outB[(size_t)m * N + n] = f2bf(v);
      }
    }
  }
}

// ---------------- Flash causal attention (WMMA, online softmax) ----------------
// grid = B*H*(T/128); 8 waves/block, each wave owns one 16-row q tile; 32-key blocks.
__global__ __launch_bounds__(256)
void flash_attn_k(const uint16_t* __restrict__ Q, const uint16_t* __restrict__ Kg,
                  const uint16_t* __restrict__ Vg, uint16_t* __restrict__ O, float scale) {
  constexpr int T = Tx, D = Dx, HS = HSx, H = Hx;
  __shared__ __align__(16) uint16_t Ks[32 * HS];      // [key][hs]
  __shared__ __align__(16) uint16_t Vt[HS * 32];      // [hs][key] (transposed for B frags)
  __shared__ __align__(16) uint16_t PS[8][16 * 32];   // per-wave P C->A relayout scratch
  const int nqb = T / 128;
  const int bh = blockIdx.x / nqb, qblk = blockIdx.x % nqb;
  const int b = bh / H, h = bh % H;
  const int tid = threadIdx.x, wid = tid >> 5, lane = tid & 31, l15 = lane & 15;
  const int half8 = (lane < 16) ? 0 : 8;
  const int abase = half8 * 2, bbase = half8 * 2;
  const int qt = qblk * 128 + wid * 16;
  const uint16_t* qrow = Q + ((size_t)b * T + qt + l15) * D + h * HS;
  v16bf qf0 = frag(qrow + abase,      qrow + abase + 8);        // hs 0..31
  v16bf qf1 = frag(qrow + 32 + abase, qrow + 32 + abase + 8);   // hs 32..63
  v8f o[4] = {};
  float mrow[8], lrow[8];
  #pragma unroll
  for (int r = 0; r < 8; r++) { mrow[r] = -3.0e38f; lrow[r] = 0.f; }
  const int nkb = (qblk + 1) * 4;
  for (int kb = 0; kb < nkb; ++kb) {
    const int k0 = kb * 32;
    { // K tile: async global->LDS (contiguous rows); V tile: sync load + LDS transpose
      const int row = tid >> 3, kc = (tid & 7) << 3;
      async_ld_b128(&Ks[row * HS + kc], &Kg[((size_t)b * T + k0 + row) * D + h * HS + kc]);
      uint4 vv = *(const uint4*)&Vg[((size_t)b * T + k0 + row) * D + h * HS + kc];
      const uint16_t* ve = (const uint16_t*)&vv;
      #pragma unroll
      for (int e = 0; e < 8; e++) Vt[(kc + e) * 32 + row] = ve[e];
      if (kb + 1 < nkb)
        __builtin_prefetch(&Kg[((size_t)b * T + k0 + 32 + row) * D + h * HS + kc], 0, 0);
    }
    wait_async0();
    __syncthreads();
    if (k0 < qt + 16) {   // causal: this wave needs keys < qt+16
      v8f s0 = {}, s1 = {};
      {
        const uint16_t* kr0 = &Ks[l15 * HS];
        const uint16_t* kr1 = &Ks[(16 + l15) * HS];
        v16bf b00 = frag(kr0 + bbase,      kr0 + bbase + 8);
        v16bf b01 = frag(kr0 + 32 + bbase, kr0 + 32 + bbase + 8);
        v16bf b10 = frag(kr1 + bbase,      kr1 + bbase + 8);
        v16bf b11 = frag(kr1 + 32 + bbase, kr1 + 32 + bbase + 8);
        s0 = wmma_bf16(qf0, b00, s0); s0 = wmma_bf16(qf1, b01, s0);
        s1 = wmma_bf16(qf0, b10, s1); s1 = wmma_bf16(qf1, b11, s1);
      }
      uint16_t* ps = &PS[wid][0];
      #pragma unroll
      for (int r = 0; r < 8; r++) {
        const int m = qt + r + half8;
        float x0 = s0[r] * scale, x1 = s1[r] * scale;
        if (k0 + l15 > m)      x0 = -3.0e38f;
        if (k0 + 16 + l15 > m) x1 = -3.0e38f;
        float mx = fmaxf(x0, x1);
        #pragma unroll
        for (int off = 8; off; off >>= 1) mx = fmaxf(mx, __shfl_xor(mx, off, 32));
        const float mnew = fmaxf(mrow[r], mx);
        const float corr = __expf(mrow[r] - mnew);
        const float e0 = __expf(x0 - mnew), e1 = __expf(x1 - mnew);
        float rs = e0 + e1;
        #pragma unroll
        for (int off = 8; off; off >>= 1) rs += __shfl_xor(rs, off, 32);
        lrow[r] = lrow[r] * corr + rs;
        mrow[r] = mnew;
        #pragma unroll
        for (int j = 0; j < 4; j++) o[j][r] *= corr;
        ps[(r + half8) * 32 + l15]      = f2bf(e0);
        ps[(r + half8) * 32 + 16 + l15] = f2bf(e1);
      }
      wait_ds0();   // P stores visible before A-frag reads (wave-private scratch)
      const uint16_t* pr = &PS[wid][l15 * 32];
      v16bf pf = frag(pr + abase, pr + abase + 8);
      #pragma unroll
      for (int j = 0; j < 4; j++) {
        const uint16_t* vr = &Vt[(j * 16 + l15) * 32];
        v16bf vf = frag(vr + bbase, vr + bbase + 8);
        o[j] = wmma_bf16(pf, vf, o[j]);
      }
    }
    __syncthreads();
  }
  #pragma unroll
  for (int j = 0; j < 4; j++) {
    const int n = h * HS + j * 16 + l15;
    #pragma unroll
    for (int r = 0; r < 8; r++) {
      const int m = qt + r + half8;
      O[((size_t)b * T + m) * D + n] = f2bf(o[j][r] / lrow[r]);
    }
  }
}

// ---------------- Compressing cross-attention ----------------
// out[r,hs] = sum_t softmax_r(q[t]·k_pos[r] * sc, mask r<=t)[r] * v[t,hs]
// grid = B*H; K_pos slice (256x64 bf16) resident in LDS; wave w owns r in [w*32, w*32+32).
__global__ __launch_bounds__(256)
void cross_attn_k(const uint16_t* __restrict__ Q2, const uint16_t* __restrict__ Kpg,
                  const uint16_t* __restrict__ V2, uint16_t* __restrict__ O, float scale) {
  constexpr int T = Tx, TR = TRx, D = Dx, HS = HSx, H = Hx;
  __shared__ __align__(16) uint16_t Kp[TR * HS];     // 32KB, whole head, loaded once
  __shared__ __align__(16) uint16_t Qs[32 * HS];     // q tile [t][hs]
  __shared__ __align__(16) uint16_t Vt[HS * 32];     // v tile transposed [hs][t]
  __shared__ __align__(16) uint16_t PS[8][32 * 32];  // per-wave P^T scratch [r][t]
  __shared__ float pmax[8][32], psum[8][32], rowM[32], rowInv[32];
  const int b = blockIdx.x / H, h = blockIdx.x % H;
  const int tid = threadIdx.x, wid = tid >> 5, lane = tid & 31, l15 = lane & 15;
  const int half8 = (lane < 16) ? 0 : 8;
  const int abase = half8 * 2, bbase = half8 * 2;
  const int r0 = wid * 32;
  for (int c = tid; c < TR * HS / 8; c += 256) {
    const int row = c >> 3, kc = (c & 7) << 3;
    async_ld_b128(&Kp[row * HS + kc], &Kpg[(size_t)row * D + h * HS + kc]);
  }
  v8f acc[2][4] = {};
  wait_async0();
  __syncthreads();
  for (int tb = 0; tb < T / 32; ++tb) {
    const int t0 = tb * 32;
    {
      const int row = tid >> 3, kc = (tid & 7) << 3;
      async_ld_b128(&Qs[row * HS + kc], &Q2[((size_t)b * T + t0 + row) * D + h * HS + kc]);
      uint4 vv = *(const uint4*)&V2[((size_t)b * T + t0 + row) * D + h * HS + kc];
      const uint16_t* ve = (const uint16_t*)&vv;
      #pragma unroll
      for (int e = 0; e < 8; e++) Vt[(kc + e) * 32 + row] = ve[e];
    }
    wait_async0();
    __syncthreads();
    // scores chunk [t 32][r 32] for this wave's r-range
    v16bf qa[2][2];
    #pragma unroll
    for (int ti = 0; ti < 2; ti++) {
      const uint16_t* qr = &Qs[(ti * 16 + l15) * HS];
      qa[ti][0] = frag(qr + abase,      qr + abase + 8);
      qa[ti][1] = frag(qr + 32 + abase, qr + 32 + abase + 8);
    }
    v8f s[2][2] = {};
    #pragma unroll
    for (int rj = 0; rj < 2; rj++) {
      const uint16_t* kr = &Kp[(r0 + rj * 16 + l15) * HS];
      v16bf kb0 = frag(kr + bbase,      kr + bbase + 8);
      v16bf kb1 = frag(kr + 32 + bbase, kr + 32 + bbase + 8);
      #pragma unroll
      for (int ti = 0; ti < 2; ti++) {
        s[ti][rj] = wmma_bf16(qa[ti][0], kb0, s[ti][rj]);
        s[ti][rj] = wmma_bf16(qa[ti][1], kb1, s[ti][rj]);
      }
    }
    // scale + mask + per-t-row partial max (softmax is over the full r=256 axis)
    #pragma unroll
    for (int ti = 0; ti < 2; ti++) {
      #pragma unroll
      for (int r = 0; r < 8; r++) {
        const int t = t0 + ti * 16 + r + half8;
        float x0 = s[ti][0][r] * scale;
        float x1 = s[ti][1][r] * scale;
        if (r0 + l15 > t)      x0 = -3.0e38f;
        if (r0 + 16 + l15 > t) x1 = -3.0e38f;
        s[ti][0][r] = x0; s[ti][1][r] = x1;
        float mx = fmaxf(x0, x1);
        #pragma unroll
        for (int off = 8; off; off >>= 1) mx = fmaxf(mx, __shfl_xor(mx, off, 32));
        if (l15 == 0) pmax[wid][ti * 16 + r + half8] = mx;
      }
    }
    __syncthreads();
    if (tid < 32) {
      float m = -3.0e38f;
      #pragma unroll
      for (int w = 0; w < 8; w++) m = fmaxf(m, pmax[w][tid]);
      rowM[tid] = m;
    }
    __syncthreads();
    #pragma unroll
    for (int ti = 0; ti < 2; ti++) {
      #pragma unroll
      for (int r = 0; r < 8; r++) {
        const int tl = ti * 16 + r + half8;
        const float mf = rowM[tl];
        float e0 = __expf(s[ti][0][r] - mf);
        float e1 = __expf(s[ti][1][r] - mf);
        s[ti][0][r] = e0; s[ti][1][r] = e1;
        float rs = e0 + e1;
        #pragma unroll
        for (int off = 8; off; off >>= 1) rs += __shfl_xor(rs, off, 32);
        if (l15 == 0) psum[wid][tl] = rs;
      }
    }
    __syncthreads();
    if (tid < 32) {
      float ssum = 0.f;
      #pragma unroll
      for (int w = 0; w < 8; w++) ssum += psum[w][tid];
      rowInv[tid] = 1.f / ssum;
    }
    __syncthreads();
    // normalize + write P^T into per-wave scratch [r_local][t_local]
    uint16_t* ps = &PS[wid][0];
    #pragma unroll
    for (int ti = 0; ti < 2; ti++)
      #pragma unroll
      for (int rj = 0; rj < 2; rj++)
        #pragma unroll
        for (int r = 0; r < 8; r++) {
          const int tl = ti * 16 + r + half8;
          const int rl = rj * 16 + l15;
          ps[rl * 32 + tl] = f2bf(s[ti][rj][r] * rowInv[tl]);
        }
    wait_ds0();
    // OUT[r, hs] += P^T[r, t-block] * V[t-block, hs]
    #pragma unroll
    for (int ri = 0; ri < 2; ri++) {
      const uint16_t* pr = &PS[wid][(ri * 16 + l15) * 32];
      v16bf pa = frag(pr + abase, pr + abase + 8);
      #pragma unroll
      for (int j = 0; j < 4; j++) {
        const uint16_t* vr = &Vt[(j * 16 + l15) * 32];
        v16bf vf = frag(vr + bbase, vr + bbase + 8);
        acc[ri][j] = wmma_bf16(pa, vf, acc[ri][j]);
      }
    }
    __syncthreads();
  }
  #pragma unroll
  for (int ri = 0; ri < 2; ri++) {
    #pragma unroll
    for (int j = 0; j < 4; j++) {
      const int n = h * HS + j * 16 + l15;
      #pragma unroll
      for (int r = 0; r < 8; r++) {
        const int rg = r0 + ri * 16 + r + half8;
        O[((size_t)b * TR + rg) * D + n] = f2bf(acc[ri][j][r]);
      }
    }
  }
}

// ---------------- host-side orchestration ----------------
extern "C" void kernel_launch(void* const* d_in, const int* in_sizes, int n_in,
                              void* d_out, int out_size, void* d_ws, size_t ws_size,
                              hipStream_t stream) {
  (void)in_sizes; (void)n_in; (void)out_size; (void)ws_size;
  const float* x     = (const float*)d_in[0];
  const float* pos   = (const float*)d_in[1];
  const float* bln1g = (const float*)d_in[2];
  const float* bln1b = (const float*)d_in[3];
  const float* bwq   = (const float*)d_in[4];
  const float* bwk   = (const float*)d_in[5];
  const float* bwv   = (const float*)d_in[6];
  const float* bwo   = (const float*)d_in[7];
  const float* bbo   = (const float*)d_in[8];
  const float* bln2g = (const float*)d_in[9];
  const float* bln2b = (const float*)d_in[10];
  const float* bw1   = (const float*)d_in[11];
  const float* bb1   = (const float*)d_in[12];
  const float* bw2   = (const float*)d_in[13];
  const float* bb2   = (const float*)d_in[14];
  const float* ln1g  = (const float*)d_in[15];
  const float* ln1b  = (const float*)d_in[16];
  const float* cawq  = (const float*)d_in[17];
  const float* cawk  = (const float*)d_in[18];
  const float* cawv  = (const float*)d_in[19];
  const float* cawo  = (const float*)d_in[20];
  const float* cabo  = (const float*)d_in[21];
  const float* ln2g  = (const float*)d_in[22];
  const float* ln2b  = (const float*)d_in[23];
  const float* ffw1  = (const float*)d_in[24];
  const float* ffb1  = (const float*)d_in[25];
  const float* ffw2  = (const float*)d_in[26];
  const float* ffb2  = (const float*)d_in[27];
  float* out = (float*)d_out;

  // workspace carve-out (~237 MB with reuse)
  char* wsb = (char*)d_ws;
  size_t off = 0;
  auto au16 = [&](size_t n) -> uint16_t* { uint16_t* p = (uint16_t*)(wsb + off);
    off += ((n * 2) + 255) & ~(size_t)255; return p; };
  auto af32 = [&](size_t n) -> float* { float* p = (float*)(wsb + off);
    off += ((n * 4) + 255) & ~(size_t)255; return p; };

  uint16_t* Wqt = au16((size_t)Dx * Dx);     // bf16 weights, [N,K]
  uint16_t* Wkt = au16((size_t)Dx * Dx);
  uint16_t* Wvt = au16((size_t)Dx * Dx);
  uint16_t* Wot = au16((size_t)Dx * Dx);
  uint16_t* W1t = au16((size_t)FFx * Dx);
  uint16_t* W2t = au16((size_t)Dx * FFx);
  uint16_t* CQt = au16((size_t)Dx * Dx);
  uint16_t* CKt = au16((size_t)Dx * Dx);
  uint16_t* CVt = au16((size_t)Dx * Dx);
  uint16_t* CWot= au16((size_t)Dx * Dx);
  uint16_t* FW1t= au16((size_t)FFx * Dx);
  uint16_t* FW2t= au16((size_t)Dx * FFx);
  uint16_t* Hbf = au16((size_t)BT * Dx);     // LN1 out; reused: attention out; cross-attn out
  uint16_t* Qbf = au16((size_t)BT * Dx);     // q; reused: h2 (LN2 out), h1c (cross LN out)
  uint16_t* Kbf = au16((size_t)BT * Dx);     // k; reused: q2
  uint16_t* Vbf = au16((size_t)BT * Dx);     // v; reused: v2
  float*    Yf  = af32((size_t)BT * Dx);     // y; in-place y2; in-place x2 (= x + y2)
  uint16_t* F1bf= au16((size_t)BT * FFx);    // inner FFN hidden (64MB)
  uint16_t* PosB= au16((size_t)TRx * Dx);    // pos_emb bf16 (GEMM A)
  uint16_t* KKbf= au16((size_t)TRx * Dx);    // k_pos projection
  float*    O1f = af32((size_t)BTR * Dx);    // cross out-proj (fp32, pre final FFN)
  uint16_t* H3bf= au16((size_t)BTR * Dx);    // LN of O1
  uint16_t* FC1bf=au16((size_t)BTR * FFx);   // final FFN hidden

  const float sc = 1.0f / 32.0f;  // c**-0.5, c = D = 1024 (reference scales by full dim)

  // ---- weight prep (bf16, pre-transposed) ----
  cvt_qkv_w<<<Dx * Dx / 256, 256, 0, stream>>>(bwq, Wqt);
  cvt_qkv_w<<<Dx * Dx / 256, 256, 0, stream>>>(bwk, Wkt);
  cvt_qkv_w<<<Dx * Dx / 256, 256, 0, stream>>>(bwv, Wvt);
  cvt_qkv_w<<<Dx * Dx / 256, 256, 0, stream>>>(cawq, CQt);
  cvt_qkv_w<<<Dx * Dx / 256, 256, 0, stream>>>(cawk, CKt);
  cvt_qkv_w<<<Dx * Dx / 256, 256, 0, stream>>>(cawv, CVt);
  transpose_w<<<dim3(Dx / 32, Dx / 32), 256, 0, stream>>>(bwo,  Wot,  Dx, Dx);
  transpose_w<<<dim3(Dx / 32, FFx / 32), 256, 0, stream>>>(bw1, W1t,  Dx, FFx);
  transpose_w<<<dim3(FFx / 32, Dx / 32), 256, 0, stream>>>(bw2, W2t,  FFx, Dx);
  transpose_w<<<dim3(Dx / 32, Dx / 32), 256, 0, stream>>>(cawo, CWot, Dx, Dx);
  transpose_w<<<dim3(Dx / 32, FFx / 32), 256, 0, stream>>>(ffw1, FW1t, Dx, FFx);
  transpose_w<<<dim3(FFx / 32, Dx / 32), 256, 0, stream>>>(ffw2, FW2t, FFx, Dx);
  cvt_f2bf<<<TRx * Dx / 256, 256, 0, stream>>>(pos, PosB, TRx * Dx);

  // ---- inner block ----
  layernorm_bf16_k<<<BT, 256, 0, stream>>>(x, bln1g, bln1b, Hbf);
  gemm_bf16_k<false,false,false,false,true><<<(BT/128)*(Dx/128), 256, 0, stream>>>(
      Hbf, Wqt, nullptr, nullptr, nullptr, Qbf, BT, Dx, Dx);
  gemm_bf16_k<false,false,false,false,true><<<(BT/128)*(Dx/128), 256, 0, stream>>>(
      Hbf, Wkt, nullptr, nullptr, nullptr, Kbf, BT, Dx, Dx);
  gemm_bf16_k<false,false,false,false,true><<<(BT/128)*(Dx/128), 256, 0, stream>>>(
      Hbf, Wvt, nullptr, nullptr, nullptr, Vbf, BT, Dx, Dx);
  flash_attn_k<<<Bx * Hx * (Tx/128), 256, 0, stream>>>(Qbf, Kbf, Vbf, Hbf, sc);  // att -> Hbf
  gemm_bf16_k<false,true,true,true,false><<<(BT/128)*(Dx/128), 256, 0, stream>>>(
      Hbf, Wot, bbo, x, Yf, nullptr, BT, Dx, Dx);                 // y = x + att@wo + bo
  layernorm_bf16_k<<<BT, 256, 0, stream>>>(Yf, bln2g, bln2b, Qbf);    // h2 -> Qbf
  gemm_bf16_k<true,true,false,false,true><<<(BT/128)*(FFx/128), 256, 0, stream>>>(
      Qbf, W1t, bb1, nullptr, nullptr, F1bf, BT, FFx, Dx);        // relu(h2@w1+b1)
  gemm_bf16_k<false,true,true,true,false><<<(BT/128)*(Dx/128), 256, 0, stream>>>(
      F1bf, W2t, bb2, Yf, Yf, nullptr, BT, Dx, FFx);              // y2 = y + .. (in place)
  add_inplace<<<BT * Dx / 256, 256, 0, stream>>>(Yf, x, BT * Dx);     // x2 = x + y2

  // ---- compressing cross-attention ----
  layernorm_bf16_k<<<BT, 256, 0, stream>>>(Yf, ln1g, ln1b, Qbf);      // h1 -> Qbf
  gemm_bf16_k<false,false,false,false,true><<<(BT/128)*(Dx/128), 256, 0, stream>>>(
      Qbf, CQt, nullptr, nullptr, nullptr, Kbf, BT, Dx, Dx);      // q2 -> Kbf
  gemm_bf16_k<false,false,false,false,true><<<(BT/128)*(Dx/128), 256, 0, stream>>>(
      Qbf, CVt, nullptr, nullptr, nullptr, Vbf, BT, Dx, Dx);      // v2 -> Vbf
  gemm_bf16_k<false,false,false,false,true><<<(TRx/128)*(Dx/128), 256, 0, stream>>>(
      PosB, CKt, nullptr, nullptr, nullptr, KKbf, TRx, Dx, Dx);   // k_pos
  cross_attn_k<<<Bx * Hx, 256, 0, stream>>>(Kbf, KKbf, Vbf, Hbf, sc); // out -> Hbf
  gemm_bf16_k<false,true,false,true,false><<<(BTR/128)*(Dx/128), 256, 0, stream>>>(
      Hbf, CWot, cabo, nullptr, O1f, nullptr, BTR, Dx, Dx);       // o1 = out@wo + bo

  // ---- final FFN ----
  layernorm_bf16_k<<<BTR, 256, 0, stream>>>(O1f, ln2g, ln2b, H3bf);
  gemm_bf16_k<true,true,false,false,true><<<(BTR/128)*(FFx/128), 256, 0, stream>>>(
      H3bf, FW1t, ffb1, nullptr, nullptr, FC1bf, BTR, FFx, Dx);
  gemm_bf16_k<false,true,true,true,false><<<(BTR/128)*(Dx/128), 256, 0, stream>>>(
      FC1bf, FW2t, ffb2, O1f, out, nullptr, BTR, Dx, FFx);        // d_out = o1 + ffn
}
